// mls_gat_classifier_71416716198365
// MI455X (gfx1250) — matrix operand order repaired and checked by reading, not compiled, not run on hardware
//
#include <hip/hip_runtime.h>
#include <hip/hip_bf16.h>
#include <stdint.h>

// ---------------- CDNA5 (gfx1250) wave32 WMMA GAT implementation ----------------
// Shapes (fixed by the reference): B=16, F=C=32, N=200, T=48, S=3, ORDER=2
// All heavy math runs through v_wmma_f32_16x16x32_bf16 with hand-built fragments.

typedef __attribute__((ext_vector_type(16))) __bf16 v16bf;
typedef __attribute__((ext_vector_type(8)))  float  v8f;

#define GAT_NEG_INF (-9.0e15f)
#define GAT_ALPHA   (0.2f)

constexpr int B_   = 16;
constexpr int F_   = 32;   // in/out feature width of each GAT layer
constexpr int N_   = 200;  // nodes
constexpr int T_   = 48;   // time
constexpr int S_   = 3;    // supports
constexpr int NT_  = N_ * T_;       // 9600 (divisible by 16)
constexpr int CH_  = 192;           // concat channels excluding raw x (6 layers * 32)
constexpr int CCAT = 224;           // 32 (x) + 192
constexpr int CO_  = 64;            // mlp out channels
constexpr int MPAD = 224;           // padded m extent for WhT (7 chunks of 32)

// --------------------------------------------------------------------------
// wa1[b,f] = sum_c W[b,f,c] * a[b,c]        (so f1 = x . wa1 exactly, fp32)
// wa2[b,f] = sum_c W[b,f,c] * a[b,32+c]
__global__ void k_wa(const float* __restrict__ W, const float* __restrict__ a,
                     float* __restrict__ wa1, float* __restrict__ wa2) {
  int b = blockIdx.x;
  int t = threadIdx.x;             // 64 threads
  int f = t & 31;
  const float* Wb = W + b * F_ * F_ + f * F_;
  const float* ab = a + b * 2 * F_ + ((t >> 5) ? F_ : 0);
  float s = 0.f;
#pragma unroll
  for (int c = 0; c < F_; ++c) s += Wb[c] * ab[c];
  if (t < 32) wa1[b * F_ + f] = s;
  else        wa2[b * F_ + f] = s;
}

// --------------------------------------------------------------------------
// WhT[b,t,c,m] = (sum_f X[b,f,n,t] * W[b,f,c]) as bf16, m=n, MPAD padded.
// Also f1[b,nt] = sum_f X[f,nt]*wa1[f], f2 likewise (exact fp32, fused).
// grid = (150, B), block = 128 (4 waves, one 16-wide nt tile each)
__global__ void k_wh(const float* __restrict__ X, long xBatchStride,
                     const float* __restrict__ W,
                     const float* __restrict__ wa1, const float* __restrict__ wa2,
                     __bf16* __restrict__ WhT,
                     float* __restrict__ f1, float* __restrict__ f2) {
  __shared__ __bf16 sW[F_ * F_];   // W[f][c] in bf16
  __shared__ float  sA[2 * F_];    // wa1 | wa2
  const int b   = blockIdx.y;
  const int tid = threadIdx.x;

  const float* Wb = W + b * F_ * F_;
  for (int i = tid; i < F_ * F_; i += 128) sW[i] = (__bf16)Wb[i];
  if (tid < F_)           sA[tid] = wa1[b * F_ + tid];
  else if (tid < 2 * F_)  sA[tid] = wa2[b * F_ + tid - F_];
  __syncthreads();

  const int wave = tid >> 5, lane = tid & 31;
  const int col  = lane & 15, half = lane >> 4;
  const int ntb  = (blockIdx.x * 4 + wave) * 16;
  const float* Xb = X + (long)b * xBatchStride;

  // B fragment: X[f, ntb+col], f = half*16 + e  (coalesced across lanes per e)
  v16bf bfrag;
  float xv[16];
#pragma unroll
  for (int e = 0; e < 16; ++e) {
    int f = half * 16 + e;
    float v = Xb[(long)f * NT_ + ntb + col];
    xv[e] = v;
    bfrag[e] = (__bf16)v;
  }

  // fused f1/f2 (exact fp32): reduce over f (this lane has half of f-range)
  float s1 = 0.f, s2 = 0.f;
#pragma unroll
  for (int e = 0; e < 16; ++e) {
    int f = half * 16 + e;
    s1 += xv[e] * sA[f];
    s2 += xv[e] * sA[F_ + f];
  }
  s1 += __shfl_xor(s1, 16, 32);
  s2 += __shfl_xor(s2, 16, 32);
  if (half == 0) {
    f1[b * NT_ + ntb + col] = s1;
    f2[b * NT_ + ntb + col] = s2;
  }

  // A fragments: A[m=c, k=f] = W[f, c] for c-blocks {0,16}
  v16bf a0, a1;
#pragma unroll
  for (int e = 0; e < 16; ++e) {
    int f = half * 16 + e;
    a0[e] = sW[f * F_ + 0  + col];
    a1[e] = sW[f * F_ + 16 + col];
  }
  v8f acc0 = {}, acc1 = {};
  acc0 = __builtin_amdgcn_wmma_f32_16x16x32_bf16(false, a0, false, bfrag, (short)0, acc0, false, false);
  acc1 = __builtin_amdgcn_wmma_f32_16x16x32_bf16(false, a1, false, bfrag, (short)0, acc1, false, false);

  // scatter to WhT[b,t,c,m] (bf16, transposed for contiguous B-loads later)
  const int nt = ntb + col;
  const int nn = nt / T_, tt = nt - nn * T_;
  __bf16* base = WhT + (long)(b * T_ + tt) * F_ * MPAD + nn;
#pragma unroll
  for (int r = 0; r < 8; ++r) {
    int c0 = r + 8 * half;
    base[(long)c0 * MPAD]        = (__bf16)acc0[r];
    base[(long)(16 + c0) * MPAD] = (__bf16)acc1[r];
  }
}

// --------------------------------------------------------------------------
// Fused masked-softmax attention + aggregation (flash style, online softmax).
// out[c,n] = sum_m softmax_m(mask(lrelu(f1[n]+f2[m]))) * Wh[c,m]  per (b,t)
// grid = (T, B), block = 256 (8 waves), each wave owns n-tiles {w, w+8}.
__global__ void k_att(const __bf16* __restrict__ WhT,
                      const float* __restrict__ f1, const float* __restrict__ f2,
                      const int* __restrict__ adjS,   // support + s*N*N
                      float* __restrict__ out,        // hcat + slot*NT
                      int actMode) {                  // 0 = elu, 1 = relu
  const int t = blockIdx.x, b = blockIdx.y;
  __shared__ float sf1[224], sf2[224];
  const int tid = threadIdx.x;
  if (tid < 224) {
    float v1 = 0.f, v2 = 0.f;
    if (tid < N_) {
      v1 = f1[b * NT_ + tid * T_ + t];
      v2 = f2[b * NT_ + tid * T_ + t];
    }
    sf1[tid] = v1; sf2[tid] = v2;
  }
  __syncthreads();

  const int wave = tid >> 5, lane = tid & 31;
  const int col  = lane & 15, half = lane >> 4;
  const int* adjb = adjS + (long)b * (S_ * N_ * N_);
  const __bf16* whtB = WhT + (long)(b * T_ + t) * F_ * MPAD;

  for (int tile = wave; tile < 13; tile += 8) {
    const int nb = tile * 16;
    const int nr = nb + col;                  // this lane's stat row (A-row)
    const int nrc = nr < N_ ? nr : N_ - 1;    // clamp for safe loads
    const float frow = sf1[nr];               // nr <= 207 < 224 (padded zeros)
    const int* adjrow = adjb + (long)nrc * N_;

    float Mx = -__builtin_inff();
    float Sm = 0.f;
    v8f acc0 = {}, acc1 = {};

    for (int mc = 0; mc < 7; ++mc) {
      const int mb = mc * 32;
      float ev[16];
      float lmax = -__builtin_inff();
#pragma unroll
      for (int e = 0; e < 16; ++e) {
        int m = mb + half * 16 + e;
        float val;
        if (m < N_) {
          float raw = frow + sf2[m];
          float le  = raw > 0.f ? raw : GAT_ALPHA * raw;
          val = (adjrow[m] > 0) ? le : GAT_NEG_INF;
        } else {
          val = -__builtin_inff();            // padded columns vanish
        }
        ev[e] = val;
        lmax = fmaxf(lmax, val);
      }
      float pmax = fmaxf(lmax, __shfl_xor(lmax, 16, 32));
      float newM = fmaxf(Mx, pmax);
      float scale = __expf(Mx - newM);        // first chunk: exp(-inf)=0

      v16bf pa;
      float lsum = 0.f;
#pragma unroll
      for (int e = 0; e < 16; ++e) {
        float p = __expf(ev[e] - newM);
        lsum += p;
        pa[e] = (__bf16)p;
      }
      Sm = Sm * scale + lsum + __shfl_xor(lsum, 16, 32);
      Mx = newM;

      // rescale accumulators: C-row (r+8*half) correction lives at lane (8*half+r)
#pragma unroll
      for (int r = 0; r < 8; ++r) {
        float cr = __shfl(scale, half * 8 + r, 32);
        acc0[r] *= cr;
        acc1[r] *= cr;
      }

      // B fragments: WhT[.., c, mb + half*16 .. +15] -> contiguous 32B loads
      const __bf16* p0 = whtB + (long)(col)      * MPAD + mb + half * 16;
      const __bf16* p1 = whtB + (long)(16 + col) * MPAD + mb + half * 16;
      v16bf b0 = *(const v16bf*)p0;
      v16bf b1 = *(const v16bf*)p1;

      acc0 = __builtin_amdgcn_wmma_f32_16x16x32_bf16(false, pa, false, b0, (short)0, acc0, false, false);
      acc1 = __builtin_amdgcn_wmma_f32_16x16x32_bf16(false, pa, false, b1, (short)0, acc1, false, false);
    }

    // normalize, activate, store: D row = n, D col = c
    const long obase = (long)b * (CH_ * NT_);
#pragma unroll
    for (int r = 0; r < 8; ++r) {
      float Sr = __shfl(Sm, half * 8 + r, 32);
      int n = nb + r + 8 * half;
      if (n < N_) {
        float inv = 1.0f / Sr;
        float h0 = acc0[r] * inv;
        float h1 = acc1[r] * inv;
        if (actMode == 0) {                    // ELU
          h0 = h0 > 0.f ? h0 : (__expf(h0) - 1.0f);
          h1 = h1 > 0.f ? h1 : (__expf(h1) - 1.0f);
        } else {                               // relu(elu(.)) == relu(.)
          h0 = fmaxf(h0, 0.f);
          h1 = fmaxf(h1, 0.f);
        }
        out[obase + (long)(col)      * NT_ + n * T_ + t] = h0;
        out[obase + (long)(16 + col) * NT_ + n * T_ + t] = h1;
      }
    }
  }
}

// --------------------------------------------------------------------------
// Final 1x1 conv: out[b,o,nt] = bias[o] + sum_{c<224} cat[c,nt] * mlp_w[o,c]
// cat channel c: c<32 -> x, else hcat. grid = (75, B), block = 256 (8 waves).
__global__ void k_mlp(const float* __restrict__ x, const float* __restrict__ hcat,
                      const float* __restrict__ mw, const float* __restrict__ mb,
                      float* __restrict__ out) {
  __shared__ __bf16 sW[CO_ * CCAT];   // 64*224 bf16 = 28 KB
  __shared__ float  sB[CO_];
  const int tid = threadIdx.x, b = blockIdx.y;
  for (int i = tid; i < CO_ * CCAT; i += 256) sW[i] = (__bf16)mw[i];
  if (tid < CO_) sB[tid] = mb[tid];
  __syncthreads();

  const int wave = tid >> 5, lane = tid & 31;
  const int col  = lane & 15, half = lane >> 4;
  const int ntb  = (blockIdx.x * 8 + wave) * 16;
  const float* xb = x    + (long)b * F_  * NT_;
  const float* hb = hcat + (long)b * CH_ * NT_;

  v8f acc[4] = {};
  for (int kc = 0; kc < 7; ++kc) {
    const int cb = kc * 32;
    v16bf bfrag;
#pragma unroll
    for (int e = 0; e < 16; ++e) {
      int c = cb + half * 16 + e;
      float v = (c < F_) ? xb[(long)c * NT_ + ntb + col]
                         : hb[(long)(c - F_) * NT_ + ntb + col];
      bfrag[e] = (__bf16)v;
    }
#pragma unroll
    for (int ot = 0; ot < 4; ++ot) {
      v16bf afrag;
#pragma unroll
      for (int e = 0; e < 16; ++e) {
        int c = cb + half * 16 + e;
        afrag[e] = sW[(ot * 16 + col) * CCAT + c];
      }
      acc[ot] = __builtin_amdgcn_wmma_f32_16x16x32_bf16(false, afrag, false, bfrag, (short)0, acc[ot], false, false);
    }
  }
#pragma unroll
  for (int ot = 0; ot < 4; ++ot) {
#pragma unroll
    for (int r = 0; r < 8; ++r) {
      int o = ot * 16 + r + 8 * half;
      out[(long)b * CO_ * NT_ + (long)o * NT_ + ntb + col] = acc[ot][r] + sB[o];
    }
  }
}

// --------------------------------------------------------------------------
extern "C" void kernel_launch(void* const* d_in, const int* in_sizes, int n_in,
                              void* d_out, int out_size, void* d_ws, size_t ws_size,
                              hipStream_t stream) {
  (void)in_sizes; (void)n_in; (void)out_size; (void)ws_size;
  const float* x   = (const float*)d_in[0];
  const int*   sup = (const int*)  d_in[1];
  const float* W1  = (const float*)d_in[2];
  const float* a1  = (const float*)d_in[3];
  const float* WK  = (const float*)d_in[4];
  const float* aK  = (const float*)d_in[5];
  const float* mw  = (const float*)d_in[6];
  const float* mbv = (const float*)d_in[7];
  float* out = (float*)d_out;

  char* ws = (char*)d_ws;
  size_t off = 0;
  auto alloc = [&](size_t bytes) {
    size_t o = off; off += (bytes + 255) & ~(size_t)255; return o;
  };
  float*  hcat = (float*) (ws + alloc((size_t)B_ * CH_ * NT_ * 4));       // ~118 MB
  __bf16* wht  = (__bf16*)(ws + alloc((size_t)B_ * T_ * F_ * MPAD * 2));  // ~11 MB
  float*  f1   = (float*) (ws + alloc((size_t)B_ * NT_ * 4));
  float*  f2   = (float*) (ws + alloc((size_t)B_ * NT_ * 4));
  float*  wa1  = (float*) (ws + alloc((size_t)B_ * F_ * 4));
  float*  wa2  = (float*) (ws + alloc((size_t)B_ * F_ * 4));

  // zero WhT once: padding columns (m>=200) must stay finite (p==0 kills them)
  hipMemsetAsync(wht, 0, (size_t)B_ * T_ * F_ * MPAD * 2, stream);

  for (int s = 0; s < S_; ++s) {
    const int* adjS = sup + (size_t)s * N_ * N_;

    // ---- layer A: x1 = elu(gat(x)), channel slot s*64 ----
    const float* Ws = W1 + (size_t)s * B_ * F_ * F_;
    const float* as = a1 + (size_t)s * B_ * 2 * F_;
    k_wa <<<dim3(B_),       64,  0, stream>>>(Ws, as, wa1, wa2);
    k_wh <<<dim3(150, B_), 128, 0, stream>>>(x, (long)F_ * NT_, Ws, wa1, wa2, wht, f1, f2);
    k_att<<<dim3(T_,  B_), 256, 0, stream>>>(wht, f1, f2, adjS,
                                             hcat + (size_t)(s * 64) * NT_, /*elu*/0);

    // ---- layer B: x2 = relu(gat(x1)), channel slot s*64+32 ----
    const float* Wk = WK + (size_t)s * B_ * F_ * F_;
    const float* ak = aK + (size_t)s * B_ * 2 * F_;
    k_wa <<<dim3(B_),       64,  0, stream>>>(Wk, ak, wa1, wa2);
    k_wh <<<dim3(150, B_), 128, 0, stream>>>(hcat + (size_t)(s * 64) * NT_,
                                             (long)CH_ * NT_, Wk, wa1, wa2, wht, f1, f2);
    k_att<<<dim3(T_,  B_), 256, 0, stream>>>(wht, f1, f2, adjS,
                                             hcat + (size_t)(s * 64 + 32) * NT_, /*relu*/1);
  }

  // ---- final 1x1 conv over concat [x | hcat] ----
  k_mlp<<<dim3(75, B_), 256, 0, stream>>>(x, hcat, mw, mbv, out);
}
